// AimNet_41558103556141
// MI455X (gfx1250) — matrix operand order repaired and checked by reading, not compiled, not run on hardware
//
#include <hip/hip_runtime.h>

// CDNA5 / gfx1250 fused AimNet forward, round 4.
// d_ws staging (one-time, deterministic): W2/P1 -> f16 (natural layout == WMMA-B layout),
// softmax(q)*(1-I) -> f16 in WMMA-A swizzled order, W1/b1 -> f16 in WMMA-A swizzled order.
// Main kernel: 16 batch rows / workgroup, 8 wave32 waves, all intermediates in LDS;
// all WMMA operands are aligned vector loads; phase-1 activation in packed f16.

typedef __attribute__((ext_vector_type(16))) _Float16 v16h;
typedef __attribute__((ext_vector_type(4)))  _Float16 v4h;
typedef __attribute__((ext_vector_type(8)))  float    v8f;

#define WMMA_F16(a, b, c) \
    __builtin_amdgcn_wmma_f32_16x16x32_f16(false, (a), false, (b), (short)0, (c), false, false)

constexpr int Bsz = 8192;
constexpr int C   = 128;
constexpr int E   = 64;
constexpr int BT  = 16;       // batch rows per workgroup
constexpr int NPAD   = 136;   // v_lds row stride in halves -> 16B aligned B-operand chunks
constexpr int CTXPAD = 1040;  // ctx_lds row stride in halves -> 32B aligned A-operand chunks

constexpr int V_HALVES   = BT * E * NPAD;   // 1024 * 136
constexpr int CTX_HALVES = 16 * CTXPAD;     // 16 * 1040
constexpr size_t LDS_BYTES = (size_t)(V_HALVES + CTX_HALVES) * 2;  // 311,808 B

constexpr size_t WT_ELEMS  = (size_t)C * E * E;            // 524288 per weight tensor
constexpr size_t WS_W2H    = 0;                            // f16 W2   (1 MB)
constexpr size_t WS_P1H    = WT_ELEMS;                     // f16 P1   (1 MB)
constexpr size_t WS_WSM    = 2 * WT_ELEMS;                 // f16 softmax table (32 KB)
constexpr size_t WS_W1H    = WS_WSM + (size_t)C * C;       // f16 W1, A-swizzled (16 KB)
constexpr size_t WS_B1H    = WS_W1H + (size_t)C * E;       // f16 b1, A-swizzled (16 KB)

// Position of K value k (0..31) inside a 16-bit WMMA A operand:
// returns hlA*16 + element index; contiguous order: K = 16*hlA + element.
__device__ __forceinline__ int pos32(int k) {
    return k;
}
__device__ __forceinline__ int pos64(int k) {           // K in 0..63 -> two A operands
    return ((k >> 5) << 5) + pos32(k & 31);
}

// ---------------- Prep 1: W2, P1 -> f16 (natural [c][f][e] layout) ------------------
__global__ __launch_bounds__(256)
void prep_weights(const float* __restrict__ W2, const float* __restrict__ P1,
                  _Float16* __restrict__ W2h, _Float16* __restrict__ P1h)
{
    const size_t i = ((size_t)blockIdx.x * 256 + threadIdx.x) * 4;
    if (i + 3 < WT_ELEMS) {
        float4 a = *(const float4*)(W2 + i);
        v4h ha = { (_Float16)a.x, (_Float16)a.y, (_Float16)a.z, (_Float16)a.w };
        *(v4h*)(W2h + i) = ha;
        float4 b = *(const float4*)(P1 + i);
        v4h hb = { (_Float16)b.x, (_Float16)b.y, (_Float16)b.z, (_Float16)b.w };
        *(v4h*)(P1h + i) = hb;
    }
}

// ---------------- Prep 2: softmax(q) * (1-I) -> f16, A-swizzled ---------------------
__global__ __launch_bounds__(256)
void prep_softmax(const float* __restrict__ q, _Float16* __restrict__ wsm)
{
    const int tid  = threadIdx.x;
    const int wave = tid >> 5;
    const int lane = tid & 31;
    const int hl   = lane >> 4;
    const int lh   = lane & 15;
    const int c    = blockIdx.x * 16 + wave * 2 + hl;

    float qv[8];
#pragma unroll
    for (int i = 0; i < 8; ++i) qv[i] = q[(size_t)c * C + lh + 16 * i];
    float m = qv[0];
#pragma unroll
    for (int i = 1; i < 8; ++i) m = fmaxf(m, qv[i]);
#pragma unroll
    for (int s2 = 1; s2 < 16; s2 <<= 1) m = fmaxf(m, __shfl_xor(m, s2, 32));
    float s = 0.f;
#pragma unroll
    for (int i = 0; i < 8; ++i) { qv[i] = __expf(qv[i] - m); s += qv[i]; }
#pragma unroll
    for (int s2 = 1; s2 < 16; s2 <<= 1) s += __shfl_xor(s, s2, 32);
    const float rs = 1.f / s;
#pragma unroll
    for (int i = 0; i < 8; ++i) {
        const int n = lh + 16 * i;
        float w = qv[i] * rs;
        if (n == c) w = 0.f;                         // zero diagonal after softmax
        wsm[(size_t)c * C + ((n >> 5) << 5) + pos32(n & 31)] = (_Float16)w;
    }
}

// ---------------- Prep 3: W1, b1 -> f16, A-swizzled [c][pos64(e)] -------------------
__global__ __launch_bounds__(256)
void prep_w1b1(const float* __restrict__ W1, const float* __restrict__ b1,
               _Float16* __restrict__ W1h, _Float16* __restrict__ B1h)
{
    const int i = blockIdx.x * 256 + threadIdx.x;   // 32 blocks -> 8192 elements
    const int c = i >> 6;
    const int k = i & 63;
    const int dst = c * E + pos64(k);
    W1h[dst] = (_Float16)W1[i];
    B1h[dst] = (_Float16)b1[i];
}

// ---------------- Main fused kernel -------------------------------------------------
__global__ __launch_bounds__(256)
void aimnet_fused(const float* __restrict__ samples,
                  const _Float16* __restrict__ W1h, const _Float16* __restrict__ B1h,
                  const _Float16* __restrict__ W2h, const float* __restrict__ b2,
                  const _Float16* __restrict__ wsm,
                  const _Float16* __restrict__ P1h, const float* __restrict__ pb1,
                  const float* __restrict__ P2, const float* __restrict__ pb2,
                  float* __restrict__ out)
{
    extern __shared__ _Float16 smem[];
    _Float16* v_lds   = smem;              // [b*64+e][n] normalized values
    _Float16* ctx_lds = smem + V_HALVES;   // [c_local][b*64 + Apos(e)]  (A-swizzled)

    const int tid  = threadIdx.x;
    const int wave = tid >> 5;
    const int lane = tid & 31;
    const int hl   = lane >> 4;
    const int lh   = lane & 15;
    const int b0   = blockIdx.x * BT;

    const v16h zero16 = (v16h)(_Float16)0.f;

    // ---- Phase 1: per-column value MLP + L2 norm -> v_lds ----
#pragma unroll 1
    for (int ci = 0; ci < 16; ++ci) {
        const int c = wave * 16 + ci;
        if (ci + 1 < 16)
            __builtin_prefetch(W2h + ((size_t)(c + 1) * E) * E, 0, 1);

        // A operands: h = relu(s*W1 + b1) in packed f16 (operands pre-swizzled)
        const _Float16 sh = (_Float16)samples[(size_t)(b0 + lh) * C + c];  // row b = lh
        v16h sv;
#pragma unroll
        for (int e = 0; e < 16; ++e) sv[e] = sh;

        const _Float16* w1p = W1h + c * E + hl * 16;
        const _Float16* b1p = B1h + c * E + hl * 16;
        v16h a0 = *(const v16h*)(w1p)      * sv + *(const v16h*)(b1p);
        v16h a1 = *(const v16h*)(w1p + 32) * sv + *(const v16h*)(b1p + 32);
        a0 = __builtin_elementwise_max(a0, zero16);   // relu -> v_pk_max_num_f16
        a1 = __builtin_elementwise_max(a1, zero16);

        v8f acc[4];
#pragma unroll
        for (int t = 0; t < 4; ++t) {
            const int f = t * 16 + lh;
            const _Float16* w2r = W2h + (((size_t)c * E + f) << 6) + 16 * hl;
            v16h bk0 = *(const v16h*)(w2r);        // K = 16*hl + 0..15
            v16h bk1 = *(const v16h*)(w2r + 32);   // K = 32 + 16*hl + 0..15
            v8f z = {};
            z = WMMA_F16(a0, bk0, z);
            z = WMMA_F16(a1, bk1, z);
            const float bias = b2[c * E + f];
#pragma unroll
            for (int r = 0; r < 8; ++r) z[r] += bias;
            acc[t] = z;
        }

        float inv[8];
#pragma unroll
        for (int r = 0; r < 8; ++r) {
            float ss = 0.f;
#pragma unroll
            for (int t = 0; t < 4; ++t) ss += acc[t][r] * acc[t][r];
#pragma unroll
            for (int m = 1; m < 16; m <<= 1) ss += __shfl_xor(ss, m, 32);
            inv[r] = 1.f / fmaxf(sqrtf(ss), 1e-12f);
        }
#pragma unroll
        for (int t = 0; t < 4; ++t)
#pragma unroll
            for (int r = 0; r < 8; ++r) {
                const int b  = r + 8 * hl;
                const int be = b * E + t * 16 + lh;
                v_lds[be * NPAD + c] = (_Float16)(acc[t][r] * inv[r]);
            }
    }
    __syncthreads();

    // ---- Phases 2+3: attention mix + projection, 16 columns at a time ----
#pragma unroll 1
    for (int ct = 0; ct < 8; ++ct) {
        // Phase 2: ctx[c_local,(b,e)] = sum_n w[c,n]*v[b,n,e]; M=16,K=128,N=1024
        v16h aw[4];   // A operands from pre-swizzled softmax table (row = c_local = lh)
#pragma unroll
        for (int kt = 0; kt < 4; ++kt)
            aw[kt] = *(const v16h*)(wsm + (size_t)(ct * 16 + lh) * C + kt * 32 + hl * 16);

#pragma unroll 1
        for (int j = 0; j < 8; ++j) {
            const int nt = wave * 8 + j;
            const int be = nt * 16 + lh;
            v8f z = {};
#pragma unroll
            for (int kt = 0; kt < 4; ++kt) {
                v16h bv = *(const v16h*)(v_lds + be * NPAD + kt * 32 + 16 * hl);
                z = WMMA_F16(aw[kt], bv, z);
            }
            // store into ctx in A-swizzled order for phase 3
            const int eloc = (j & 3) * 16 + lh;
            const int b    = wave * 2 + (j >> 2);
            const int base = b * E + pos64(eloc);
#pragma unroll
            for (int r = 0; r < 8; ++r)
                ctx_lds[(r + 8 * hl) * CTXPAD + base] = (_Float16)z[r];
        }
        __syncthreads();

        // Phase 3: per-column projection MLP + P2 dot; 2 columns per wave
#pragma unroll 1
        for (int cc = 0; cc < 2; ++cc) {
            const int cl = wave + 8 * cc;
            const int c  = ct * 16 + cl;
            __builtin_prefetch(P1h + ((size_t)c * E) * E, 0, 1);

            // A operands: one aligned 32B LDS load each (pre-swizzled ctx)
            const _Float16* crow = ctx_lds + cl * CTXPAD + lh * E + hl * 16;
            v16h a0 = *(const v16h*)(crow);
            v16h a1 = *(const v16h*)(crow + 32);

            float partial[8];
#pragma unroll
            for (int r = 0; r < 8; ++r) partial[r] = 0.f;

#pragma unroll
            for (int t = 0; t < 4; ++t) {
                const int f = t * 16 + lh;
                const _Float16* p1r = P1h + (((size_t)c * E + f) << 6) + 16 * hl;
                v16h bk0 = *(const v16h*)(p1r);
                v16h bk1 = *(const v16h*)(p1r + 32);
                v8f z = {};
                z = WMMA_F16(a0, bk0, z);
                z = WMMA_F16(a1, bk1, z);
                const float bias = pb1[c * E + f];
                const float p2v  = P2[c * E + f];
#pragma unroll
                for (int r = 0; r < 8; ++r) {
                    const float h2 = fmaxf(0.f, z[r] + bias);
                    partial[r] += h2 * p2v;
                }
            }

            float myout = 0.f;
#pragma unroll
            for (int r = 0; r < 8; ++r) {
                float v = partial[r];
#pragma unroll
                for (int m = 1; m < 16; m <<= 1) v += __shfl_xor(v, m, 32);
                if (lh == r) myout = v + pb2[c];
            }
            if (lh < 8)
                out[(size_t)(b0 + lh + 8 * hl) * C + c] = myout;
        }
        __syncthreads();
    }
}

extern "C" void kernel_launch(void* const* d_in, const int* in_sizes, int n_in,
                              void* d_out, int out_size, void* d_ws, size_t ws_size,
                              hipStream_t stream) {
    (void)in_sizes; (void)n_in; (void)out_size; (void)ws_size;
    const float* samples = (const float*)d_in[0];
    const float* W1  = (const float*)d_in[1];
    const float* b1  = (const float*)d_in[2];
    const float* W2  = (const float*)d_in[3];
    const float* b2  = (const float*)d_in[4];
    const float* q   = (const float*)d_in[5];
    const float* P1  = (const float*)d_in[6];
    const float* pb1 = (const float*)d_in[7];
    const float* P2  = (const float*)d_in[8];
    const float* pb2 = (const float*)d_in[9];

    _Float16* wsbase = (_Float16*)d_ws;
    _Float16* W2h = wsbase + WS_W2H;
    _Float16* P1h = wsbase + WS_P1H;
    _Float16* wsm = wsbase + WS_WSM;
    _Float16* W1h = wsbase + WS_W1H;
    _Float16* B1h = wsbase + WS_B1H;

    // deterministic staging every call
    prep_weights<<<dim3((unsigned)(WT_ELEMS / 4 / 256)), dim3(256), 0, stream>>>(W2, P1, W2h, P1h);
    prep_softmax<<<dim3(8), dim3(256), 0, stream>>>(q, wsm);
    prep_w1b1<<<dim3(32), dim3(256), 0, stream>>>(W1, b1, W1h, B1h);

    aimnet_fused<<<dim3(Bsz / BT), dim3(256), LDS_BYTES, stream>>>(
        samples, W1h, B1h, W2h, b2, wsm, P1h, pb1, P2, pb2, (float*)d_out);
}
